// BLTModel_35003983462517
// MI455X (gfx1250) — matrix operand order repaired and checked by reading, not compiled, not run on hardware
//
#include <hip/hip_runtime.h>
#include <hip/hip_bf16.h>
#include <cstdint>
#include <cstddef>

#define DEV __device__ __forceinline__

namespace {

constexpr int Bq = 4, Pn = 256, Tn = 1024, Kn = 8;
constexpr int Dm = 512, FFm = 2048, NHn = 8, HDn = 64;
constexpr int LEn = 2, LDn = 2, VOC = 32000, BUK = 50000;
constexpr float NEGI = -1e9f;

typedef __attribute__((ext_vector_type(16))) _Float16 v16h;
typedef __attribute__((ext_vector_type(8)))  _Float16 v8h;   // 16B LDS vector load
typedef __attribute__((ext_vector_type(2)))  _Float16 v2h;   // packed b32 store
typedef __attribute__((ext_vector_type(8)))  float    v8f;

union FragA { v16h v; v8h q[2]; _Float16 h[16]; };
union FragC { v8f  v; float f[8]; };

// ---------------- wave32 reductions ----------------
DEV float waveReduceSum(float v) {
#pragma unroll
  for (int m = 16; m > 0; m >>= 1) v += __shfl_xor(v, m, 32);
  return v;
}
DEV float waveReduceMax(float v) {
#pragma unroll
  for (int m = 16; m > 0; m >>= 1) v = fmaxf(v, __shfl_xor(v, m, 32));
  return v;
}

// ====================================================================
// Generic batched GEMM:  C[g] = act(A[g] x B[g] + bias)
// A: [M,K] f32 row-major, B: [K,N] f32 row-major. f16 WMMA, f32 accum.
// Block: 256 threads = 8 waves; block tile 128x64; wave tile 32x32
// (4 WMMAs per chunk, each A/B fragment reused twice -> 2.0 LDS
// loads per WMMA). A staged row-major (frag = contiguous 16B runs);
// B staged TRANSPOSED (Bs[n][k]) so each lane's 16 K-halfs are two
// contiguous 16B runs (K = 8h.., 16+8h..).
// outHeads: write C as [B, NH, Tlen, HD] instead of [M, N] (QKV split).
// ====================================================================
__launch_bounds__(256)
__global__ void wmma_gemm(const float* __restrict__ A, const float* __restrict__ Bm,
                          const float* __restrict__ bias, float* __restrict__ C,
                          int M, int N, int Kd,
                          long long sA, long long sB, long long sC,
                          int act, int outHeads, int Tlen)
{
  constexpr int LDA = 40;   // halfs per A row  (32 + 8 pad, 80B = 16B aligned)
  constexpr int LDB = 40;   // halfs per B *column* (32 + 8 pad)
  __shared__ __align__(16) _Float16 As[128 * LDA];
  __shared__ __align__(16) _Float16 Bs[64 * LDB];   // [n][k] transposed

  const int g = blockIdx.z;
  A  += (long long)g * sA;
  Bm += (long long)g * sB;
  C  += (long long)g * sC;

  const int rowBase = blockIdx.y * 128;
  const int colBase = blockIdx.x * 64;
  const int tid  = threadIdx.x;
  const int wave = tid >> 5;
  const int lane = tid & 31;
  const int wm   = wave >> 1;       // 0..3 -> 32-row slice
  const int wn   = wave & 1;        // 0..1 -> 32-col slice
  const int m16  = lane & 15;
  const int hh   = lane >> 4;

  FragC acc[2][2];
#pragma unroll
  for (int i = 0; i < 2; ++i)
#pragma unroll
    for (int j = 0; j < 2; ++j)
#pragma unroll
      for (int r = 0; r < 8; ++r) acc[i][j].f[r] = 0.f;

  for (int k0 = 0; k0 < Kd; k0 += 32) {
    // ---- stage A tile 128x32 (row-major f16); 16 elems/thread ----
    {
      const int r = tid >> 1;                // 0..127
      const int c = (tid & 1) * 16;          // 0,16
      const float* src = A + (long long)(rowBase + r) * Kd + k0 + c;
#pragma unroll
      for (int j = 0; j < 16; ++j) As[r * LDA + c + j] = (_Float16)src[j];
    }
    // ---- stage B tile 32x64 transposed -> Bs[n][k]; 8 elems/thread ----
    {
      const int kk = (tid >> 4) * 2;         // 0,2,..,30 (even)
      const int nb = (tid & 15) * 4;         // 0,4,..,60
      const float* s0 = Bm + (long long)(k0 + kk) * N + colBase + nb;
      const float* s1 = s0 + N;
#pragma unroll
      for (int j = 0; j < 4; ++j) {
        v2h pk;
        pk.x = (_Float16)s0[j];
        pk.y = (_Float16)s1[j];
        *(v2h*)&Bs[(nb + j) * LDB + kk] = pk;   // 4B store, kk even
      }
    }
    if (k0 + 32 < Kd) {   // prefetch next K tile (near-temporal)
      __builtin_prefetch(A + (long long)(rowBase + (tid >> 1)) * Kd + k0 + 32, 0, 3);
      __builtin_prefetch(Bm + (long long)(k0 + 32 + ((tid >> 4) * 2)) * N + colBase, 0, 3);
    }
    __syncthreads();

    // ---- fragments: contiguous 16B LDS vector loads ----
    FragA af0, af1, bf0, bf1;
    {
      const v8h* Ar0 = (const v8h*)&As[(wm * 32 + m16) * LDA];
      const v8h* Ar1 = (const v8h*)&As[(wm * 32 + 16 + m16) * LDA];
      af0.q[0] = Ar0[hh];          // K = 8h .. 8h+7
      af0.q[1] = Ar0[2 + hh];      // K = 16+8h .. 16+8h+7
      af1.q[0] = Ar1[hh];
      af1.q[1] = Ar1[2 + hh];
    }
    {
      const v8h* Br0 = (const v8h*)&Bs[(wn * 32 + m16) * LDB];
      const v8h* Br1 = (const v8h*)&Bs[(wn * 32 + 16 + m16) * LDB];
      bf0.q[0] = Br0[hh];          // K = 8h .. 8h+7
      bf0.q[1] = Br0[2 + hh];      // K = 16+8h .. 16+8h+7
      bf1.q[0] = Br1[hh];
      bf1.q[1] = Br1[2 + hh];
    }
    acc[0][0].v = __builtin_amdgcn_wmma_f32_16x16x32_f16(false, af0.v, false, bf0.v,
                                                         (short)0, acc[0][0].v, false, false);
    acc[0][1].v = __builtin_amdgcn_wmma_f32_16x16x32_f16(false, af0.v, false, bf1.v,
                                                         (short)0, acc[0][1].v, false, false);
    acc[1][0].v = __builtin_amdgcn_wmma_f32_16x16x32_f16(false, af1.v, false, bf0.v,
                                                         (short)0, acc[1][0].v, false, false);
    acc[1][1].v = __builtin_amdgcn_wmma_f32_16x16x32_f16(false, af1.v, false, bf1.v,
                                                         (short)0, acc[1][1].v, false, false);
    __syncthreads();
  }

  // epilogue: C/D layout -> VGPR r holds M = r + 8*half, N = lane&15
#pragma unroll
  for (int ti = 0; ti < 2; ++ti) {
#pragma unroll
    for (int tj = 0; tj < 2; ++tj) {
      FragC& a = acc[ti][tj];
#pragma unroll
      for (int r = 0; r < 8; ++r) {
        const int gm = rowBase + wm * 32 + ti * 16 + r + 8 * hh;
        const int gn = colBase + wn * 32 + tj * 16 + m16;
        float v = a.f[r];
        if (bias) v += bias[gn];
        if (act)  v = fmaxf(v, 0.f);
        if (outHeads) {
          const int bb = gm / Tlen, tt = gm % Tlen;
          const int h2 = gn >> 6,  dd = gn & 63;
          C[(((long long)(bb * NHn + h2)) * Tlen + tt) * HDn + dd] = v;
        } else {
          C[(long long)gm * N + gn] = v;
        }
      }
    }
  }
}

// ====================================================================
// Attention scores:  S[g,q,k] = scale * Q[g,q,:] . K[g,k,:] + mask bias
// Q,K: [G, T, HD] f32 (G = B*NH).  HD = 64 (two K-chunks of 32).
// Both tiles staged row-major; since B = K^T, the B-fragment reads are
// K-contiguous runs of a single Ks row -> 16B vector loads.
// ====================================================================
__launch_bounds__(256)
__global__ void wmma_scores(const float* __restrict__ Q, const float* __restrict__ Kv,
                            float* __restrict__ S, int Tq, int Tk,
                            const unsigned char* __restrict__ keymask,
                            int causal, float scale)
{
  constexpr int LDQ = 72;   // 64 + 8 pad halfs, 144B = 16B aligned
  __shared__ __align__(16) _Float16 Qs[64 * LDQ];
  __shared__ __align__(16) _Float16 Ks[64 * LDQ];

  const int g = blockIdx.z;            // b*NH + h
  const int b = g / NHn;
  Q  += (long long)g * Tq * HDn;
  Kv += (long long)g * Tk * HDn;
  S  += (long long)g * Tq * Tk;

  const int qBase = blockIdx.y * 64;
  const int kBase = blockIdx.x * 64;
  const int tid = threadIdx.x, wave = tid >> 5, lane = tid & 31;
  const int wm = wave >> 1, wn = wave & 1, m16 = lane & 15, hh = lane >> 4;

  // stage 64x64 Q rows and K rows (full HD depth)
  {
    const int r = tid >> 2;            // 0..63
    const int c = (tid & 3) * 16;      // 0,16,32,48
    const float* qs = Q  + (long long)(qBase + r) * HDn + c;
    const float* ks = Kv + (long long)(kBase + r) * HDn + c;
#pragma unroll
    for (int j = 0; j < 16; ++j) Qs[r * LDQ + c + j] = (_Float16)qs[j];
#pragma unroll
    for (int j = 0; j < 16; ++j) Ks[r * LDQ + c + j] = (_Float16)ks[j];
  }
  __syncthreads();

  FragC acc0, acc1;
#pragma unroll
  for (int i = 0; i < 8; ++i) { acc0.f[i] = 0.f; acc1.f[i] = 0.f; }

#pragma unroll
  for (int kc = 0; kc < 2; ++kc) {
    FragA af, bf0, bf1;
    {
      const v8h* Qr = (const v8h*)&Qs[(wm * 16 + m16) * LDQ + kc * 32];
      af.q[0] = Qr[hh];
      af.q[1] = Qr[2 + hh];
    }
    {
      const v8h* Kr0 = (const v8h*)&Ks[(wn * 32 + m16) * LDQ + kc * 32];
      const v8h* Kr1 = (const v8h*)&Ks[(wn * 32 + 16 + m16) * LDQ + kc * 32];
      bf0.q[0] = Kr0[hh];
      bf0.q[1] = Kr0[2 + hh];
      bf1.q[0] = Kr1[hh];
      bf1.q[1] = Kr1[2 + hh];
    }
    acc0.v = __builtin_amdgcn_wmma_f32_16x16x32_f16(false, af.v, false, bf0.v,
                                                    (short)0, acc0.v, false, false);
    acc1.v = __builtin_amdgcn_wmma_f32_16x16x32_f16(false, af.v, false, bf1.v,
                                                    (short)0, acc1.v, false, false);
  }

#pragma unroll
  for (int t = 0; t < 2; ++t) {
    FragC& acc = t ? acc1 : acc0;
#pragma unroll
    for (int r = 0; r < 8; ++r) {
      const int gq = qBase + wm * 16 + r + 8 * hh;
      const int gk = kBase + wn * 32 + t * 16 + m16;
      float v = acc.f[r] * scale;
      if (keymask && !keymask[b * Tk + gk]) v += NEGI;
      if (causal && gk > gq)                v += NEGI;
      S[(long long)gq * Tk + gk] = v;
    }
  }
}

// ---------------- softmax over last dim (in place) ----------------
__launch_bounds__(256)
__global__ void softmax_rows(float* __restrict__ S, int cols)
{
  __shared__ float sm[8];
  const long long base = (long long)blockIdx.x * cols;
  const int tid = threadIdx.x;

  float mx = -3.4e38f;
  for (int c = tid; c < cols; c += 256) mx = fmaxf(mx, S[base + c]);
  mx = waveReduceMax(mx);
  if ((tid & 31) == 0) sm[tid >> 5] = mx;
  __syncthreads();
  float m = sm[0];
#pragma unroll
  for (int i = 1; i < 8; ++i) m = fmaxf(m, sm[i]);
  __syncthreads();

  float sum = 0.f;
  for (int c = tid; c < cols; c += 256) {
    float e = __expf(S[base + c] - m);
    S[base + c] = e;
    sum += e;
  }
  sum = waveReduceSum(sum);
  if ((tid & 31) == 0) sm[tid >> 5] = sum;
  __syncthreads();
  float tot = 0.f;
#pragma unroll
  for (int i = 0; i < 8; ++i) tot += sm[i];
  const float inv = 1.0f / tot;
  for (int c = tid; c < cols; c += 256) S[base + c] *= inv;
}

// ---------------- out = LayerNorm(A + Bm) * g + beta ----------------
__launch_bounds__(256)
__global__ void ln_residual(const float* __restrict__ A, const float* __restrict__ Bm,
                            const float* __restrict__ g, const float* __restrict__ bt,
                            float* __restrict__ out)
{
  __shared__ float sm[8];
  const long long base = (long long)blockIdx.x * Dm;
  const int tid = threadIdx.x;

  const float x0 = A[base + tid]       + Bm[base + tid];
  const float x1 = A[base + tid + 256] + Bm[base + tid + 256];

  float s = waveReduceSum(x0 + x1);
  if ((tid & 31) == 0) sm[tid >> 5] = s;
  __syncthreads();
  float tot = 0.f;
#pragma unroll
  for (int i = 0; i < 8; ++i) tot += sm[i];
  const float mu = tot / (float)Dm;
  __syncthreads();

  const float c0 = x0 - mu, c1 = x1 - mu;
  float v = waveReduceSum(c0 * c0 + c1 * c1);
  if ((tid & 31) == 0) sm[tid >> 5] = v;
  __syncthreads();
  float var = 0.f;
#pragma unroll
  for (int i = 0; i < 8; ++i) var += sm[i];
  var /= (float)Dm;
  const float rs = rsqrtf(var + 1e-5f);

  out[base + tid]       = c0 * rs * g[tid]       + bt[tid];
  out[base + tid + 256] = c1 * rs * g[tid + 256] + bt[tid + 256];
}

// ---------------- n-gram gather-sum patch embedding ----------------
__launch_bounds__(256)
__global__ void patch_embed(const int* __restrict__ ngram,
                            const unsigned char* __restrict__ pmask,
                            const float* __restrict__ tables,
                            const float* __restrict__ ppos,
                            float* __restrict__ X)
{
  __shared__ int sid[3 * Kn];
  const int bp  = blockIdx.x;              // b*P + p
  const int p   = bp % Pn;
  const int tid = threadIdx.x;
  if (tid < 3 * Kn) sid[tid] = ngram[bp * (3 * Kn) + tid];
  __syncthreads();

  const bool on = pmask[bp] != 0;
  float a0 = 0.f, a1 = 0.f;
#pragma unroll
  for (int n = 0; n < 3; ++n) {
    const float* tab = tables + (long long)n * BUK * Dm;
#pragma unroll
    for (int j = 0; j < Kn; ++j) {
      const float* row = tab + (long long)sid[n * Kn + j] * Dm;
      a0 += row[tid];
      a1 += row[tid + 256];
    }
  }
  if (!on) { a0 = 0.f; a1 = 0.f; }
  X[(long long)bp * Dm + tid]       = a0 + ppos[p * Dm + tid];
  X[(long long)bp * Dm + tid + 256] = a1 + ppos[p * Dm + tid + 256];
}

// ---------------- token embedding + positional ----------------
__launch_bounds__(256)
__global__ void token_embed(const int* __restrict__ ids,
                            const float* __restrict__ emb,
                            const float* __restrict__ pos,
                            float* __restrict__ Y)
{
  const int bt  = blockIdx.x;              // b*T + t
  const int t   = bt % Tn;
  const int tid = threadIdx.x;
  const long long id = ids[bt];
  Y[(long long)bt * Dm + tid]       = emb[id * Dm + tid]       + pos[t * Dm + tid];
  Y[(long long)bt * Dm + tid + 256] = emb[id * Dm + tid + 256] + pos[t * Dm + tid + 256];
}

// ---------------- [B,NH,T,HD] -> [B*T, D] ----------------
__launch_bounds__(256)
__global__ void merge_heads(const float* __restrict__ OH, float* __restrict__ OM, int Tlen)
{
  const long long i = (long long)blockIdx.x * 256 + threadIdx.x;
  const long long tot = (long long)Bq * Tlen * Dm;
  if (i >= tot) return;
  const int c  = (int)(i % Dm);
  const long long bt = i / Dm;
  const int t = (int)(bt % Tlen);
  const int b = (int)(bt / Tlen);
  const int h = c >> 6, d = c & 63;
  OM[i] = OH[(((long long)(b * NHn + h)) * Tlen + t) * HDn + d];
}

} // anonymous namespace

// ====================================================================
// Host orchestration
// ====================================================================
extern "C" void kernel_launch(void* const* d_in, const int* in_sizes, int n_in,
                              void* d_out, int out_size, void* d_ws, size_t ws_size,
                              hipStream_t stream)
{
  (void)in_sizes; (void)n_in; (void)out_size; (void)ws_size;

  const int*           ngram_ids   = (const int*)d_in[0];
  const unsigned char* patch_mask  = (const unsigned char*)d_in[1];
  const int*           target_ids  = (const int*)d_in[2];
  const unsigned char* target_mask = (const unsigned char*)d_in[3];
  const float* tables  = (const float*)d_in[4];
  const float* ppos    = (const float*)d_in[5];
  const float* temb    = (const float*)d_in[6];
  const float* tpos    = (const float*)d_in[7];
  const float* enc_W   = (const float*)d_in[8];
  const float* enc_b   = (const float*)d_in[9];
  const float* enc_W1  = (const float*)d_in[10];
  const float* enc_b1  = (const float*)d_in[11];
  const float* enc_W2  = (const float*)d_in[12];
  const float* enc_b2  = (const float*)d_in[13];
  const float* enc_lng = (const float*)d_in[14];
  const float* enc_lnb = (const float*)d_in[15];
  const float* dec_saW = (const float*)d_in[16];
  const float* dec_sab = (const float*)d_in[17];
  const float* dec_caW = (const float*)d_in[18];
  const float* dec_cab = (const float*)d_in[19];
  const float* dec_W1  = (const float*)d_in[20];
  const float* dec_b1  = (const float*)d_in[21];
  const float* dec_W2  = (const float*)d_in[22];
  const float* dec_b2  = (const float*)d_in[23];
  const float* dec_lng = (const float*)d_in[24];
  const float* dec_lnb = (const float*)d_in[25];
  const float* Wout    = (const float*)d_in[26];
  const float* bout    = (const float*)d_in[27];

  // workspace layout (persistent activations, ~61 MB)
  float* ws  = (float*)d_ws;
  float* X   = ws;                               // [B,P,D] encoder act / memory
  float* Y   = X  + (size_t)Bq * Pn * Dm;        // [B,T,D] decoder act
  float* QH  = Y  + (size_t)Bq * Tn * Dm;        // [B,NH,T,HD]
  float* KH  = QH + (size_t)Bq * Tn * Dm;
  float* VH  = KH + (size_t)Bq * Tn * Dm;
  float* OH  = VH + (size_t)Bq * Tn * Dm;
  float* OM  = OH + (size_t)Bq * Tn * Dm;        // merged heads [B*T, D]
  float* TMP = OM + (size_t)Bq * Tn * Dm;        // sublayer output [B*T, D]
  // d_out (524 MB) doubles as scratch for scores (134 MB) and FFN hidden
  // (33 MB); both are fully consumed before the final logits GEMM rewrites it.
  float* S  = (float*)d_out;
  float* Hh = (float*)d_out;

  auto gemm = [&](const float* A, const float* Bp, const float* bias, float* C,
                  int M, int N, int Kd, int G,
                  long long sA, long long sB, long long sC,
                  int act, int outHeads, int Tlen) {
    dim3 grid(N / 64, M / 128, G);
    wmma_gemm<<<grid, 256, 0, stream>>>(A, Bp, bias, C, M, N, Kd,
                                        sA, sB, sC, act, outHeads, Tlen);
  };

  // MHA: resid_out = O-proj( softmax(scale*QK^T + bias) V )
  auto attention = [&](const float* Qsrc, int Tq, const float* KVsrc, int Tk,
                       const float* W, const float* bv,
                       const unsigned char* kmask, int causal, float* resid_out) {
    const int Mq = Bq * Tq, Mk = Bq * Tk;
    gemm(Qsrc,  W + 0 * Dm * Dm, bv + 0 * Dm, QH, Mq, Dm, Dm, 1, 0, 0, 0, 0, 1, Tq);
    gemm(KVsrc, W + 1 * Dm * Dm, bv + 1 * Dm, KH, Mk, Dm, Dm, 1, 0, 0, 0, 0, 1, Tk);
    gemm(KVsrc, W + 2 * Dm * Dm, bv + 2 * Dm, VH, Mk, Dm, Dm, 1, 0, 0, 0, 0, 1, Tk);
    {
      dim3 grid(Tk / 64, Tq / 64, Bq * NHn);
      wmma_scores<<<grid, 256, 0, stream>>>(QH, KH, S, Tq, Tk, kmask, causal, 0.125f);
    }
    softmax_rows<<<Bq * NHn * Tq, 256, 0, stream>>>(S, Tk);
    gemm(S, VH, nullptr, OH, Tq, HDn, Tk, Bq * NHn,
         (long long)Tq * Tk, (long long)Tk * HDn, (long long)Tq * HDn, 0, 0, 0);
    {
      const long long tot = (long long)Bq * Tq * Dm;
      merge_heads<<<(unsigned)((tot + 255) / 256), 256, 0, stream>>>(OH, OM, Tq);
    }
    gemm(OM, W + 3 * Dm * Dm, bv + 3 * Dm, resid_out, Mq, Dm, Dm, 1, 0, 0, 0, 0, 0, 0);
  };

  auto ffn = [&](float* act, int M, const float* W1, const float* b1,
                 const float* W2, const float* b2, const float* g, const float* bb) {
    gemm(act, W1, b1, Hh, M, FFm, Dm, 1, 0, 0, 0, 1 /*relu*/, 0, 0);
    gemm(Hh, W2, b2, TMP, M, Dm, FFm, 1, 0, 0, 0, 0, 0, 0);
    ln_residual<<<M, 256, 0, stream>>>(act, TMP, g, bb, act);
  };

  // ---------------- encoder ----------------
  patch_embed<<<Bq * Pn, 256, 0, stream>>>(ngram_ids, patch_mask, tables, ppos, X);
  for (int l = 0; l < LEn; ++l) {
    const float* W  = enc_W + (size_t)l * 4 * Dm * Dm;
    const float* bv = enc_b + (size_t)l * 4 * Dm;
    attention(X, Pn, X, Pn, W, bv, patch_mask, 0, TMP);
    ln_residual<<<Bq * Pn, 256, 0, stream>>>(X, TMP,
        enc_lng + (size_t)(l * 2 + 0) * Dm, enc_lnb + (size_t)(l * 2 + 0) * Dm, X);
    ffn(X, Bq * Pn,
        enc_W1 + (size_t)l * Dm * FFm, enc_b1 + (size_t)l * FFm,
        enc_W2 + (size_t)l * FFm * Dm, enc_b2 + (size_t)l * Dm,
        enc_lng + (size_t)(l * 2 + 1) * Dm, enc_lnb + (size_t)(l * 2 + 1) * Dm);
  }
  // X now holds encoder memory [B,P,D]

  // ---------------- decoder ----------------
  token_embed<<<Bq * Tn, 256, 0, stream>>>(target_ids, temb, tpos, Y);
  for (int l = 0; l < LDn; ++l) {
    // masked self-attention
    const float* Ws = dec_saW + (size_t)l * 4 * Dm * Dm;
    const float* bs = dec_sab + (size_t)l * 4 * Dm;
    attention(Y, Tn, Y, Tn, Ws, bs, target_mask, 1 /*causal*/, TMP);
    ln_residual<<<Bq * Tn, 256, 0, stream>>>(Y, TMP,
        dec_lng + (size_t)(l * 3 + 0) * Dm, dec_lnb + (size_t)(l * 3 + 0) * Dm, Y);
    // cross-attention to encoder memory
    const float* Wc = dec_caW + (size_t)l * 4 * Dm * Dm;
    const float* bc = dec_cab + (size_t)l * 4 * Dm;
    attention(Y, Tn, X, Pn, Wc, bc, patch_mask, 0, TMP);
    ln_residual<<<Bq * Tn, 256, 0, stream>>>(Y, TMP,
        dec_lng + (size_t)(l * 3 + 1) * Dm, dec_lnb + (size_t)(l * 3 + 1) * Dm, Y);
    // FFN
    ffn(Y, Bq * Tn,
        dec_W1 + (size_t)l * Dm * FFm, dec_b1 + (size_t)l * FFm,
        dec_W2 + (size_t)l * FFm * Dm, dec_b2 + (size_t)l * Dm,
        dec_lng + (size_t)(l * 3 + 2) * Dm, dec_lnb + (size_t)(l * 3 + 2) * Dm);
  }

  // ---------------- logits: [B*T, D] x [D, VOCAB] -> d_out ----------------
  gemm(Y, Wout, bout, (float*)d_out, Bq * Tn, VOC, Dm, 1, 0, 0, 0, 0, 0, 0);
}
